// PointerNetwork_23673859736466
// MI455X (gfx1250) — matrix operand order repaired and checked by reading, not compiled, not run on hardware
//
#include <hip/hip_runtime.h>
#include <hip/hip_bf16.h>
#include <math.h>

// Pointer-network head for MI455X (gfx1250, wave32, WMMA).
// bf16 WMMA for all GEMMs (Q/K projections + per-head scores, recomputed
// flash-style), fused scatter via global f32 atomics into d_out, then a
// fused LSE+logaddexp pass that stages each 128KB logits row in LDS
// (320KB/WGP on CDNA5) so logits are read from HBM exactly once.

typedef __attribute__((ext_vector_type(16))) __bf16 v16bf;
typedef __attribute__((ext_vector_type(8)))  float  v8f;

#define DDIM   512
#define NHEAD  8
#define HDIM   64
#define LVOC   32000
#define SRC_N  4096
#define TGT_N  2048
#define LEXT   36096
#define NEGC   (-1e9f)

// ---------- bf16 A/B fragment K-offsets (16x16x32 bf16 WMMA, wave32) ----------
// A (16xK, row=lane%16): lanes<16 get K pairs {0..7,16..23}; lanes>=16 {8..15,24..31}
__device__ __forceinline__ int kbase_a(int i, int half) {
    return (i < 4 ? 2 * i : 16 + 2 * (i - 4)) + 8 * half;
}
// B (Kx16, col=lane%16): lanes<16 get K pairs {0..7,16..23}; lanes>=16 {8..15,24..31}
__device__ __forceinline__ int kbase_b(int i, int half) {
    return (i < 4 ? 2 * i : 16 + 2 * (i - 4)) + 8 * half;
}

__device__ __forceinline__ float logsigf(float z) {
    float l = log1pf(__expf(-fabsf(z)));
    return (z < 0.f ? z : 0.f) - l;
}

// ---------------------------------------------------------------- zero output
__global__ void zero_kernel(float4* __restrict__ p, size_t n4) {
    size_t i  = (size_t)blockIdx.x * blockDim.x + threadIdx.x;
    size_t st = (size_t)gridDim.x * blockDim.x;
    float4 z = {0.f, 0.f, 0.f, 0.f};
    for (; i < n4; i += st) p[i] = z;
}

// ------------------------------------------------------------ f32 -> bf16 cvt
__global__ void cvt_bf16(const float* __restrict__ in, __bf16* __restrict__ out, int n) {
    int i  = blockIdx.x * blockDim.x + threadIdx.x;
    int st = gridDim.x * blockDim.x;
    for (; i < n; i += st) out[i] = (__bf16)in[i];
}

// ------------------------------------------- f32 -> bf16 cvt with transpose
__global__ void cvt_bf16_t(const float* __restrict__ w, __bf16* __restrict__ wt,
                           int rows, int cols) {
    int i = blockIdx.x * blockDim.x + threadIdx.x;
    if (i < rows * cols) {
        int r = i / cols, c = i % cols;
        wt[c * rows + r] = (__bf16)w[i];
    }
}

// ------------------------------------------------- C = A(MxK) * BT(NxK)^T, bf16
__global__ void gemm_abt_bf16(const __bf16* __restrict__ A, const __bf16* __restrict__ BT,
                              __bf16* __restrict__ C, int N, int K) {
    int m0 = blockIdx.x * 16, n0 = blockIdx.y * 16;
    int lane = threadIdx.x & 31, half = lane >> 4, r = lane & 15;
    const __bf16* arow = A + (size_t)(m0 + r) * K;
    const __bf16* brow = BT + (size_t)(n0 + r) * K;
    v8f acc = {};
    for (int k0 = 0; k0 < K; k0 += 32) {
        v16bf a, b;
#pragma unroll
        for (int i = 0; i < 8; ++i) {
            int ka = k0 + kbase_a(i, half);
            int kb = k0 + kbase_b(i, half);
            a[2 * i] = arow[ka]; a[2 * i + 1] = arow[ka + 1];
            b[2 * i] = brow[kb]; b[2 * i + 1] = brow[kb + 1];
        }
        acc = __builtin_amdgcn_wmma_f32_16x16x32_bf16(false, a, false, b,
                                                      (short)0, acc, false, false);
    }
#pragma unroll
    for (int v = 0; v < 8; ++v) {
        int row = v + 8 * half;
        C[(size_t)(m0 + row) * N + n0 + r] = (__bf16)acc[v];
    }
}

// ---------------------- per-(tgt-tile, head) online softmax stats over SRC ----
__global__ void attn_stats(const __bf16* __restrict__ Q, const __bf16* __restrict__ Km,
                           float* __restrict__ m_ws, float* __restrict__ s_ws) {
    int m0 = blockIdx.x * 16;
    int h  = blockIdx.y;
    int lane = threadIdx.x & 31, half = lane >> 4, r = lane & 15;
    const __bf16* arow = Q + (size_t)(m0 + r) * DDIM + h * HDIM;

    v16bf a0, a1;                                  // loop-invariant A fragments
#pragma unroll
    for (int i = 0; i < 8; ++i) {
        int ka = kbase_a(i, half);
        a0[2 * i] = arow[ka];      a0[2 * i + 1] = arow[ka + 1];
        a1[2 * i] = arow[32 + ka]; a1[2 * i + 1] = arow[32 + ka + 1];
    }

    float m[8], s[8];
#pragma unroll
    for (int v = 0; v < 8; ++v) { m[v] = -1e30f; s[v] = 0.f; }

    for (int s0 = 0; s0 < SRC_N; s0 += 16) {
        const __bf16* brow = Km + (size_t)(s0 + r) * DDIM + h * HDIM;
        v16bf b0, b1;
#pragma unroll
        for (int i = 0; i < 8; ++i) {
            int kb = kbase_b(i, half);
            b0[2 * i] = brow[kb];      b0[2 * i + 1] = brow[kb + 1];
            b1[2 * i] = brow[32 + kb]; b1[2 * i + 1] = brow[32 + kb + 1];
        }
        v8f acc = {};
        acc = __builtin_amdgcn_wmma_f32_16x16x32_bf16(false, a0, false, b0, (short)0, acc, false, false);
        acc = __builtin_amdgcn_wmma_f32_16x16x32_bf16(false, a1, false, b1, (short)0, acc, false, false);
#pragma unroll
        for (int v = 0; v < 8; ++v) {
            float x  = acc[v] * 0.125f;            // 1/sqrt(64)
            float mn = fmaxf(m[v], x);
            s[v] = s[v] * __expf(m[v] - mn) + __expf(x - mn);
            m[v] = mn;
        }
    }
    for (int off = 8; off >= 1; off >>= 1) {       // merge within 16-lane halves
#pragma unroll
        for (int v = 0; v < 8; ++v) {
            float mo = __shfl_xor(m[v], off, 32);
            float so = __shfl_xor(s[v], off, 32);
            float mn = fmaxf(m[v], mo);
            s[v] = s[v] * __expf(m[v] - mn) + so * __expf(mo - mn);
            m[v] = mn;
        }
    }
    if (r == 0) {
#pragma unroll
        for (int v = 0; v < 8; ++v) {
            int row = m0 + v + 8 * half;
            m_ws[h * TGT_N + row] = m[v];
            s_ws[h * TGT_N + row] = s[v];
        }
    }
}

// -------- recompute scores, normalize, head-average, scatter into d_out ------
__global__ void attn_scatter(const __bf16* __restrict__ Q, const __bf16* __restrict__ Km,
                             const float* __restrict__ m_ws, const float* __restrict__ s_ws,
                             const int* __restrict__ ids, float* __restrict__ out) {
    int m0 = blockIdx.x * 16, s0 = blockIdx.y * 16;
    int lane = threadIdx.x & 31, half = lane >> 4, r = lane & 15;

    float attn[8];
#pragma unroll
    for (int v = 0; v < 8; ++v) attn[v] = 0.f;

    for (int h = 0; h < NHEAD; ++h) {
        const __bf16* arow = Q  + (size_t)(m0 + r) * DDIM + h * HDIM;
        const __bf16* brow = Km + (size_t)(s0 + r) * DDIM + h * HDIM;
        v16bf a0, a1, b0, b1;
#pragma unroll
        for (int i = 0; i < 8; ++i) {
            int ka = kbase_a(i, half), kb = kbase_b(i, half);
            a0[2 * i] = arow[ka];      a0[2 * i + 1] = arow[ka + 1];
            a1[2 * i] = arow[32 + ka]; a1[2 * i + 1] = arow[32 + ka + 1];
            b0[2 * i] = brow[kb];      b0[2 * i + 1] = brow[kb + 1];
            b1[2 * i] = brow[32 + kb]; b1[2 * i + 1] = brow[32 + kb + 1];
        }
        v8f acc = {};
        acc = __builtin_amdgcn_wmma_f32_16x16x32_bf16(false, a0, false, b0, (short)0, acc, false, false);
        acc = __builtin_amdgcn_wmma_f32_16x16x32_bf16(false, a1, false, b1, (short)0, acc, false, false);
#pragma unroll
        for (int v = 0; v < 8; ++v) {
            int row = m0 + v + 8 * half;
            float mh = m_ws[h * TGT_N + row];
            float sh = s_ws[h * TGT_N + row];
            attn[v] += __expf(acc[v] * 0.125f - mh) / sh;
        }
    }
    int id = ids[s0 + r];
#pragma unroll
    for (int v = 0; v < 8; ++v) {
        int row = m0 + v + 8 * half;
        atomicAdd(out + (size_t)row * LEXT + id, attn[v] * 0.125f);  // mean over heads
    }
}

// --------------------------- z = tgt @ w_lin + b; log_sigmoid gates ----------
__global__ void gate_kernel(const float* __restrict__ tgt, const float* __restrict__ w,
                            const float* __restrict__ b, float* __restrict__ lc,
                            float* __restrict__ lk) {
    int wv = threadIdx.x >> 5, lane = threadIdx.x & 31;
    int row = blockIdx.x * (blockDim.x >> 5) + wv;
    if (row >= TGT_N) return;
    const float* tr = tgt + (size_t)row * DDIM;
    float sum = 0.f;
    for (int i = lane; i < DDIM; i += 32) sum += tr[i] * w[i];
    for (int off = 16; off >= 1; off >>= 1) sum += __shfl_xor(sum, off, 32);
    if (lane == 0) {
        float z = sum + b[0];
        lc[row] = logsigf(z);    // log(copy_prob)
        lk[row] = logsigf(-z);   // log(1 - copy_prob)
    }
}

// ----- fused per-row: stage logits row in LDS, LSE, then in-place logaddexp --
// out[t,j] currently holds ext_attn; rewrite with
//   logaddexp(log_softmax(logits)+log_keep, log(ext_attn)+log_copy)
__global__ void dec_final_kernel(const float* __restrict__ logits,
                                 const float* __restrict__ lc, const float* __restrict__ lk,
                                 float* __restrict__ out) {
    extern __shared__ float smem[];                // LVOC floats + 2*256 reduce
    float* rowbuf = smem;
    float* sm = smem + LVOC;
    float* ss = sm + 256;

    int t = blockIdx.x, tid = threadIdx.x, nt = blockDim.x;
    const float* row = logits + (size_t)t * LVOC;

    // stage row into LDS while accumulating online max/sumexp
    float m = -1e30f, s = 0.f;
    for (int j = tid; j < LVOC; j += nt) {
        float x = row[j];
        rowbuf[j] = x;
        float mn = fmaxf(m, x);
        s = s * __expf(m - mn) + __expf(x - mn);
        m = mn;
    }
    sm[tid] = m; ss[tid] = s;
    __syncthreads();
    for (int off = 128; off >= 1; off >>= 1) {
        if (tid < off) {
            float mo = sm[tid + off], so = ss[tid + off];
            float mn = fmaxf(sm[tid], mo);
            ss[tid] = ss[tid] * __expf(sm[tid] - mn) + so * __expf(mo - mn);
            sm[tid] = mn;
        }
        __syncthreads();
    }
    float lse = sm[0] + logf(ss[0]);
    float lkt = lk[t], lct = lc[t];

    float* orow = out + (size_t)t * LEXT;
    for (int j = tid; j < LEXT; j += nt) {
        float ext = orow[j];
        float a   = ((j < LVOC) ? (rowbuf[j] - lse) : NEGC) + lkt;
        float p   = ((ext > 0.f) ? logf(fmaxf(ext, 1e-38f)) : NEGC) + lct;
        float hi  = fmaxf(a, p), lo = fminf(a, p);
        orow[j] = hi + log1pf(__expf(lo - hi));
    }
}

extern "C" void kernel_launch(void* const* d_in, const int* in_sizes, int n_in,
                              void* d_out, int out_size, void* d_ws, size_t ws_size,
                              hipStream_t stream) {
    const float* logits = (const float*)d_in[0];
    const int*   ids    = (const int*)d_in[1];
    const float* src    = (const float*)d_in[2];
    const float* tgt    = (const float*)d_in[3];
    // d_in[4]: len_extended_vocab (constant LEXT)
    const float* w_lin  = (const float*)d_in[5];
    const float* b_lin  = (const float*)d_in[6];
    const float* Wq     = (const float*)d_in[7];
    const float* Wk     = (const float*)d_in[8];
    float* out = (float*)d_out;

    char* ws = (char*)d_ws;
    size_t off = 0;
    auto alloc = [&](size_t bytes) -> char* {
        char* p = ws + off;
        off += (bytes + 255) & ~(size_t)255;
        return p;
    };
    float*  lc     = (float*)alloc((size_t)TGT_N * 4);
    float*  lk     = (float*)alloc((size_t)TGT_N * 4);
    float*  m_ws   = (float*)alloc((size_t)NHEAD * TGT_N * 4);
    float*  s_ws   = (float*)alloc((size_t)NHEAD * TGT_N * 4);
    __bf16* tgt_bf = (__bf16*)alloc((size_t)TGT_N * DDIM * 2);
    __bf16* src_bf = (__bf16*)alloc((size_t)SRC_N * DDIM * 2);
    __bf16* wqt    = (__bf16*)alloc((size_t)DDIM * DDIM * 2);
    __bf16* wkt    = (__bf16*)alloc((size_t)DDIM * DDIM * 2);
    __bf16* qbf    = (__bf16*)alloc((size_t)TGT_N * DDIM * 2);
    __bf16* kbf    = (__bf16*)alloc((size_t)SRC_N * DDIM * 2);
    (void)ws_size; (void)in_sizes; (void)n_in; (void)out_size;

    // 1) d_out accumulates ext_attn -> zero it (LEXT*TGT divisible by 4)
    zero_kernel<<<4096, 256, 0, stream>>>((float4*)out, (size_t)TGT_N * LEXT / 4);

    // 2) bf16 conversions (weights transposed so GEMM is K-contiguous A*B^T)
    cvt_bf16<<<2048, 256, 0, stream>>>(tgt, tgt_bf, TGT_N * DDIM);
    cvt_bf16<<<4096, 256, 0, stream>>>(src, src_bf, SRC_N * DDIM);
    cvt_bf16_t<<<(DDIM * DDIM + 255) / 256, 256, 0, stream>>>(Wq, wqt, DDIM, DDIM);
    cvt_bf16_t<<<(DDIM * DDIM + 255) / 256, 256, 0, stream>>>(Wk, wkt, DDIM, DDIM);

    // 3) projections via WMMA: Q = tgt*Wq, K = src*Wk
    gemm_abt_bf16<<<dim3(TGT_N / 16, DDIM / 16), 32, 0, stream>>>(tgt_bf, wqt, qbf, DDIM, DDIM);
    gemm_abt_bf16<<<dim3(SRC_N / 16, DDIM / 16), 32, 0, stream>>>(src_bf, wkt, kbf, DDIM, DDIM);

    // 4) copy gates
    gate_kernel<<<TGT_N / 8, 256, 0, stream>>>(tgt, w_lin, b_lin, lc, lk);

    // 5) per-head online softmax stats, then recompute+normalize+scatter
    attn_stats<<<dim3(TGT_N / 16, NHEAD), 32, 0, stream>>>(qbf, kbf, m_ws, s_ws);
    attn_scatter<<<dim3(TGT_N / 16, SRC_N / 16), 32, 0, stream>>>(qbf, kbf, m_ws, s_ws, ids, out);

    // 6) fused LSE + logaddexp, one 128KB LDS-staged logits row per workgroup
    size_t shmem = (size_t)(LVOC + 512) * sizeof(float);
    dec_final_kernel<<<TGT_N, 256, shmem, stream>>>(logits, lc, lk, out);
}
